// PaGCN_54065048323072
// MI455X (gfx1250) — compile-verified, matrix operand order checked by reading, not run on hardware
//
#include <hip/hip_runtime.h>

// ---------------------------------------------------------------------------
// PaGCN forward for MI455X (gfx1250, wave32, WMMA bf16 16x16x32)
// ---------------------------------------------------------------------------

typedef __attribute__((ext_vector_type(16))) __bf16 v16bf;
typedef __attribute__((ext_vector_type(8)))  __bf16 v8bf;
typedef __attribute__((ext_vector_type(8)))  float  v8f;

__device__ __forceinline__ unsigned short f32_to_bf16_bits(float f) {
    unsigned int u = __float_as_uint(f);
    // round-to-nearest-even
    unsigned int r = u + 0x7FFFu + ((u >> 16) & 1u);
    return (unsigned short)(r >> 16);
}

// ---------------------------------------------------------------------------
// zero fill (grid-stride)
// ---------------------------------------------------------------------------
__global__ void zero_f32_kernel(float* p, long n) {
    long i = (long)blockIdx.x * blockDim.x + threadIdx.x;
    long stride = (long)gridDim.x * blockDim.x;
    for (; i < n; i += stride) p[i] = 0.0f;
}

// ---------------------------------------------------------------------------
// COO SpMM with optional per-source-node scaling:
//   Y[row[e], f] += vals[e] * (M ? M[col[e]] : 1) * X[col[e], f]
// one block per edge, blockDim == F (feature width), coalesced row access.
// global_prefetch_b8 warms the destination line before the fp32 L2 atomic.
// ---------------------------------------------------------------------------
__global__ void spmm_edge_kernel(const float* __restrict__ vals,
                                 const int* __restrict__ row,
                                 const int* __restrict__ col,
                                 const float* __restrict__ Mscale,  // may be null
                                 const float* __restrict__ X,
                                 float* __restrict__ Y,
                                 int F) {
    int e = blockIdx.x;
    int f = threadIdx.x;
    int r = row[e];
    int c = col[e];
    float* dst = &Y[(long)r * F + f];
    __builtin_prefetch(dst, 1, 3);              // -> global_prefetch_b8
    float coeff = vals[e];
    if (Mscale) coeff *= Mscale[c];
    float v = coeff * X[(long)c * F + f];
    __hip_atomic_fetch_add(dst, v, __ATOMIC_RELAXED, __HIP_MEMORY_SCOPE_AGENT);
}

// ---------------------------------------------------------------------------
// f32 -> bf16 staging, with optional per-row (node) scaling by AM:
//   out[i] = bf16(H[i] * (AM ? AM[i >> shiftF] : 1))
// ---------------------------------------------------------------------------
__global__ void cvt_bf16_kernel(const float* __restrict__ H,
                                const float* __restrict__ AM,   // may be null
                                unsigned short* __restrict__ out,
                                int shiftF, long total) {
    long i = (long)blockIdx.x * blockDim.x + threadIdx.x;
    if (i >= total) return;
    float s = AM ? AM[i >> shiftF] : 1.0f;
    out[i] = f32_to_bf16_bits(H[i] * s);
}

// ---------------------------------------------------------------------------
// Pack weight matrix W (K x Nout, row-major f32) into WMMA B-fragment order
// (bf16).  Fragment layout per ISA 7.12.2 (16-bit 32x16 operand, wave32):
//   lane l (<16):  elems 0..7 -> K = kt*32 + 0..7,   elems 8..15 -> K = kt*32 + 16..23
//   lane l (>=16): elems 0..7 -> K = kt*32 + 8..15,  elems 8..15 -> K = kt*32 + 24..31
//   column N = ntile*16 + (l & 15)
// storage: out[((ktile*ntiles + ntile)*32 + lane)*16 + j]   (contiguous per lane)
// ---------------------------------------------------------------------------
__global__ void pack_w_bf16_kernel(const float* __restrict__ W,
                                   unsigned short* __restrict__ out,
                                   int K, int Nout, int ntiles, int Ktiles) {
    int tid = blockIdx.x * blockDim.x + threadIdx.x;
    int total = Ktiles * ntiles * 32 * 16;
    if (tid >= total) return;
    int j     = tid & 15;
    int lane  = (tid >> 4) & 31;
    int tile  = tid >> 9;
    int ntile = tile % ntiles;
    int ktile = tile / ntiles;
    int hi    = (lane & 16) ? 8 : 0;
    int k     = ktile * 32 + ((j < 8) ? (hi + j) : (16 + hi + (j - 8)));
    int n     = ntile * 16 + (lane & 15);
    float v   = (k < K && n < Nout) ? W[(long)k * Nout + n] : 0.0f;
    out[tid]  = f32_to_bf16_bits(v);
}

// ---------------------------------------------------------------------------
// WMMA tile core: accumulate a 16x16 f32 tile over K = KTILES*32.
// All strides are compile-time constants so fragment loads become
// base-pointer + immediate-offset b128 loads (no per-iteration address VALU).
// ---------------------------------------------------------------------------
template <int NTILES, int KTILES>
__device__ __forceinline__ v8f wmma_tile_accum(const __bf16* __restrict__ Ab,
                                               const __bf16* __restrict__ Bb,
                                               int m0, int lane, int wv,
                                               int mrow, int hi) {
    // per-wave/lane base pointers; all k-step offsets are immediates
    const __bf16* pA = Ab + ((long)(m0 + mrow)) * (KTILES * 32) + hi;
    const __bf16* pB = Bb + ((long)(wv * 32 + lane)) * 16;
    v8f c = {};
#pragma unroll
    for (int kt = 0; kt < KTILES; ++kt) {
        v8bf alo = *reinterpret_cast<const v8bf*>(pA + kt * 32);
        v8bf ahi = *reinterpret_cast<const v8bf*>(pA + kt * 32 + 16);
        v16bf a = __builtin_shufflevector(alo, ahi,
                                          0,1,2,3,4,5,6,7,8,9,10,11,12,13,14,15);
        v16bf b = *reinterpret_cast<const v16bf*>(pB + (long)kt * NTILES * 512);
        c = __builtin_amdgcn_wmma_f32_16x16x32_bf16(
                false, a, false, b, (short)0, c, false, false);
    }
    return c;
}

// ---------------------------------------------------------------------------
// WMMA GEMM:  out[M x NOUT] = act(A[M x K] (bf16) @ Wpacked + bias)
// grid.x = M/16 tiles; blockDim.x = NTILES*32; wave w owns N-tile w.
// ---------------------------------------------------------------------------
template <int NTILES, int KTILES, int NOUT, bool RELU>
__global__ void gemm_wmma_kernel(const unsigned short* __restrict__ Abits,
                                 const unsigned short* __restrict__ Wbits,
                                 const float* __restrict__ bias,
                                 float* __restrict__ out,
                                 int Mrows) {
    const int lane = threadIdx.x & 31;
    const int wv   = threadIdx.x >> 5;          // N-tile index
    const int m0   = blockIdx.x * 16;
    const int mrow = lane & 15;
    const int hi   = (lane & 16) ? 8 : 0;

    const __bf16* Ab = reinterpret_cast<const __bf16*>(Abits);
    const __bf16* Bb = reinterpret_cast<const __bf16*>(Wbits);

    v8f c = wmma_tile_accum<NTILES, KTILES>(Ab, Bb, m0, lane, wv, mrow, hi);

    const int n = wv * 16 + (lane & 15);
    float bv = (n < NOUT) ? bias[n] : 0.0f;
#pragma unroll
    for (int i = 0; i < 8; ++i) {
        int r = m0 + hi + i;
        float v = c[i] + bv;
        if (RELU) v = fmaxf(v, 0.0f);
        if (n < NOUT && r < Mrows) out[(long)r * NOUT + n] = v;
    }
}

// ---------------------------------------------------------------------------
// Final GEMM (K=64, NOUT=40 padded to 3 N-tiles) fused with log_softmax.
// blockDim = 96 (3 waves).  16x48 f32 tile staged in LDS, then 16 threads
// perform the 40-class log-softmax.
// ---------------------------------------------------------------------------
template <int NTILES, int KTILES, int NOUT>
__global__ void gemm2_logsoftmax_kernel(const unsigned short* __restrict__ Abits,
                                        const unsigned short* __restrict__ Wbits,
                                        const float* __restrict__ bias,
                                        float* __restrict__ out,
                                        int Mrows) {
    __shared__ float tile[16][NTILES * 16 + 1];

    const int lane = threadIdx.x & 31;
    const int wv   = threadIdx.x >> 5;          // 0..NTILES-1
    const int m0   = blockIdx.x * 16;
    const int mrow = lane & 15;
    const int hi   = (lane & 16) ? 8 : 0;

    const __bf16* Ab = reinterpret_cast<const __bf16*>(Abits);
    const __bf16* Bb = reinterpret_cast<const __bf16*>(Wbits);

    v8f c = wmma_tile_accum<NTILES, KTILES>(Ab, Bb, m0, lane, wv, mrow, hi);

    const int nl = wv * 16 + (lane & 15);       // 0..NTILES*16-1
    float bv = (nl < NOUT) ? bias[nl] : 0.0f;
#pragma unroll
    for (int i = 0; i < 8; ++i) tile[hi + i][nl] = c[i] + bv;

    __syncthreads();

    if (threadIdx.x < 16) {
        int rl = threadIdx.x;
        int r  = m0 + rl;
        if (r < Mrows) {
            float mx = -3.402823466e38f;
            for (int k = 0; k < NOUT; ++k) mx = fmaxf(mx, tile[rl][k]);
            float s = 0.0f;
            for (int k = 0; k < NOUT; ++k) s += __expf(tile[rl][k] - mx);
            float lse = mx + __logf(s);
            long base = (long)r * NOUT;
            for (int k = 0; k < NOUT; ++k) out[base + k] = tile[rl][k] - lse;
        }
    }
}

// ---------------------------------------------------------------------------
// Host-side orchestration
// ---------------------------------------------------------------------------
extern "C" void kernel_launch(void* const* d_in, const int* in_sizes, int n_in,
                              void* d_out, int out_size, void* d_ws, size_t ws_size,
                              hipStream_t stream) {
    const float* x       = (const float*)d_in[0];
    const float* M       = (const float*)d_in[1];
    const float* AM      = (const float*)d_in[2];
    const float* adjv    = (const float*)d_in[3];
    const float* adjZv   = (const float*)d_in[4];
    const float* W0      = (const float*)d_in[5];
    const float* b0      = (const float*)d_in[6];
    const float* W1      = (const float*)d_in[7];
    const float* b1      = (const float*)d_in[8];
    const float* W2      = (const float*)d_in[9];
    const float* b2      = (const float*)d_in[10];
    const int*   row     = (const int*)d_in[11];
    const int*   col     = (const int*)d_in[12];

    const int N      = in_sizes[1];               // 50000
    const int E      = in_sizes[3];               // 800000
    const int NFEAT  = 128;
    const int NHID   = 64;
    const int NCLASS = 40;

    // ---- workspace partition (all offsets 256B aligned) ----
    char* ws = (char*)d_ws;
    size_t off = 0;
    auto carve = [&](size_t bytes) -> char* {
        char* p = ws + off;
        off += (bytes + 255) & ~(size_t)255;
        return p;
    };
    float*          hA   = (float*)carve((size_t)N * NFEAT * 4);   // spmm outputs
    float*          hB   = (float*)carve((size_t)N * NFEAT * 4);   // gemm0 out / spmm3 out
    unsigned short* abf  = (unsigned short*)carve((size_t)N * NFEAT * 2);
    float*          hC   = (float*)carve((size_t)N * NHID * 4);    // gemm1 out
    unsigned short* W0p  = (unsigned short*)carve((size_t)(NFEAT/32) * (NFEAT/16) * 512 * 2);
    unsigned short* W1p  = (unsigned short*)carve((size_t)(NFEAT/32) * (NHID/16)  * 512 * 2);
    const int n2tiles = (NCLASS + 15) / 16;                        // 3
    unsigned short* W2p  = (unsigned short*)carve((size_t)(NHID/32) * n2tiles * 512 * 2);
    (void)ws_size; (void)n_in; (void)in_sizes;

    const int mtiles = (N + 15) / 16;             // 3125

    // ---- pack weights into WMMA B-fragment layout (tiny, once per launch) ----
    {
        int t0 = (NFEAT/32) * (NFEAT/16) * 512;
        pack_w_bf16_kernel<<<(t0 + 255)/256, 256, 0, stream>>>(W0, W0p, NFEAT, NFEAT, NFEAT/16, NFEAT/32);
        int t1 = (NFEAT/32) * (NHID/16) * 512;
        pack_w_bf16_kernel<<<(t1 + 255)/256, 256, 0, stream>>>(W1, W1p, NFEAT, NHID, NHID/16, NFEAT/32);
        int t2 = (NHID/32) * n2tiles * 512;
        pack_w_bf16_kernel<<<(t2 + 255)/256, 256, 0, stream>>>(W2, W2p, NHID, NCLASS, n2tiles, NHID/32);
    }

    // ---- layer 0: h = spmm(adjZ, M*x) * AM ; h = relu(h@W0 + b0) ----
    zero_f32_kernel<<<4096, 256, 0, stream>>>(hA, (long)N * NFEAT);
    spmm_edge_kernel<<<E, NFEAT, 0, stream>>>(adjZv, row, col, M, x, hA, NFEAT);
    cvt_bf16_kernel<<<(int)(((long)N*NFEAT + 255)/256), 256, 0, stream>>>(
        hA, AM, abf, 7, (long)N * NFEAT);
    gemm_wmma_kernel<8, 4, 128, true><<<mtiles, 8*32, 0, stream>>>(
        abf, W0p, b0, hB, N);

    // ---- layer 1: h = spmm(adjZ, M*h) * AM ; h = relu(h@W1 + b1) ----
    zero_f32_kernel<<<4096, 256, 0, stream>>>(hA, (long)N * NFEAT);
    spmm_edge_kernel<<<E, NFEAT, 0, stream>>>(adjZv, row, col, M, hB, hA, NFEAT);
    cvt_bf16_kernel<<<(int)(((long)N*NFEAT + 255)/256), 256, 0, stream>>>(
        hA, AM, abf, 7, (long)N * NFEAT);
    gemm_wmma_kernel<4, 4, 64, true><<<mtiles, 4*32, 0, stream>>>(
        abf, W1p, b1, hC, N);

    // ---- layer 2: h = spmm(adj, h) ; out = log_softmax(h@W2 + b2) ----
    zero_f32_kernel<<<4096, 256, 0, stream>>>(hB, (long)N * NHID);
    spmm_edge_kernel<<<E, NHID, 0, stream>>>(adjv, row, col, nullptr, hC, hB, NHID);
    cvt_bf16_kernel<<<(int)(((long)N*NHID + 255)/256), 256, 0, stream>>>(
        hB, nullptr, abf, 6, (long)N * NHID);
    gemm2_logsoftmax_kernel<3, 2, 40><<<mtiles, 3*32, 0, stream>>>(
        abf, W2p, b2, (float*)d_out, N);
    (void)out_size;
}
